// Policy_1365799600835
// MI455X (gfx1250) — compile-verified, matrix operand order checked by reading, not run on hardware
//
#include <hip/hip_runtime.h>
#include <hip/hip_bf16.h>
#include <stdint.h>

// ---------------------------------------------------------------------------
// CDNA5 (gfx1250) fused policy network.
//  K0: weight f32->f16 prep (+ transposes/padding)
//  K1: per-WG(8 images): token scatter -> LDS box -> conv1 (WMMA im2col GEMM)
//      -> conv2 (WMMA im2col GEMM) -> flat features f16 to ws
//  K2: per-WG(16 rows): fc1 -> enc -> critic(+value) -> actor1 -> query ->
//      logits, all via v_wmma_f32_16x16x32_f16.
// ---------------------------------------------------------------------------

typedef __attribute__((ext_vector_type(16))) _Float16 v16h;
typedef __attribute__((ext_vector_type(8)))  _Float16 v8h;
typedef __attribute__((ext_vector_type(8)))  float    v8f;

union V16U { v16h v; v8h h[2]; };

__device__ __forceinline__ v8f vzero8() {
  v8f z;
#pragma unroll
  for (int i = 0; i < 8; ++i) z[i] = 0.0f;
  return z;
}

// Load a 16x32 f16 fragment (A or B) from a row-major panel (row stride ldk
// halves). ISA layout (CDNA5 ISA §7.12.2, 16-bit A 16x32):
//   lanes 0-15 : row = lane,      K = {0..7, 16..23}
//   lanes 16-31: row = lane-16,   K = {8..15, 24..31}
__device__ __forceinline__ v16h load_frag16(const _Float16* base, int ldk) {
  const int lane = threadIdx.x & 31;
  const int r  = lane & 15;
  const int kh = (lane >> 4) << 3;   // 0 or 8
  const _Float16* p = base + r * ldk + kh;
  V16U u;
  u.h[0] = *(const v8h*)(p);         // K = kh + 0..7
  u.h[1] = *(const v8h*)(p + 16);    // K = kh + 16..23
  return u.v;
}

__device__ __forceinline__ v8f wmma16(v16h a, v16h b, v8f c) {
  return __builtin_amdgcn_wmma_f32_16x16x32_f16(
      false, a, false, b, (short)0, c, false, false);
}

// ---------------------------------------------------------------------------
// Model constants
// ---------------------------------------------------------------------------
#define NB      8192
#define NTOK    200
#define NL      32        // conv input channels (layers)
#define CNN_CH  64
#define K1LEN   800       // 32*5*5
#define K2LEN   576       // 64*3*3
#define FLATW   256       // 64*2*2
#define HID     128
#define CRITH   1024
#define ACTH    512
#define EMBD    16
#define NACT    100
#define IMGS    8         // images per workgroup in conv kernel

// ws layout (in _Float16 elements)
#define OFF_W1    0            // [64][800]
#define OFF_W2    51200        // [64][576]
#define OFF_FC1   88064        // [128][256]
#define OFF_ENC   120832       // [128][128]
#define OFF_CRIT  137216       // [1024][128]
#define OFF_ACT1  268288       // [512][128]
#define OFF_ACTWT 333824       // [16][512] (transposed actor_W)
#define OFF_EMB   342016       // [112][32] (padded action_emb)
#define OFF_FEATS 345600       // [8192][256]

// ---------------------------------------------------------------------------
// K0: weight conversion / transpose / padding
// ---------------------------------------------------------------------------
__global__ void k_prep(const float* __restrict__ cnn1_w,
                       const float* __restrict__ cnn2_w,
                       const float* __restrict__ fc1_w,
                       const float* __restrict__ enc_w,
                       const float* __restrict__ crit_w,
                       const float* __restrict__ act1_w,
                       const float* __restrict__ actor_W,
                       const float* __restrict__ emb,
                       _Float16* __restrict__ ws) {
  const int tid = blockIdx.x * blockDim.x + threadIdx.x;
  const int stride = gridDim.x * blockDim.x;
  _Float16* w1h  = ws + OFF_W1;
  _Float16* w2h  = ws + OFF_W2;
  _Float16* fc1h = ws + OFF_FC1;
  _Float16* ench = ws + OFF_ENC;
  _Float16* crh  = ws + OFF_CRIT;
  _Float16* a1h  = ws + OFF_ACT1;
  _Float16* awt  = ws + OFF_ACTWT;
  _Float16* emh  = ws + OFF_EMB;
  for (int i = tid; i < 64 * K1LEN;  i += stride) w1h[i]  = (_Float16)cnn1_w[i];
  for (int i = tid; i < 64 * K2LEN;  i += stride) w2h[i]  = (_Float16)cnn2_w[i];
  for (int i = tid; i < 128 * 256;   i += stride) fc1h[i] = (_Float16)fc1_w[i];
  for (int i = tid; i < 128 * 128;   i += stride) ench[i] = (_Float16)enc_w[i];
  for (int i = tid; i < 1024 * 128;  i += stride) crh[i]  = (_Float16)crit_w[i];
  for (int i = tid; i < 512 * 128;   i += stride) a1h[i]  = (_Float16)act1_w[i];
  // actor_W[0] is [512][16] (h,e); store transposed [16][512] so B-frag rows
  // are contiguous in K.
  for (int i = tid; i < EMBD * ACTH; i += stride) {
    int n = i >> 9, k = i & 511;
    awt[i] = (_Float16)actor_W[k * EMBD + n];
  }
  // action_emb [100][16] -> padded [112][32] (zero-filled) for K=32 WMMA step.
  for (int i = tid; i < 112 * 32; i += stride) {
    int n = i >> 5, k = i & 31;
    emh[i] = (n < NACT && k < EMBD) ? (_Float16)emb[n * EMBD + k]
                                    : (_Float16)0.0f;
  }
}

// ---------------------------------------------------------------------------
// K1: scatter-encode + conv1 + conv2  (8 images / workgroup, 8 waves)
// Dynamic LDS layout (halves):
//   box   : 8*32*256 = 65536
//   apan  : 128*32   =  4096
//   c1out : 128*64   =  8192
//   a2pan : 32*32    =  1024
//   total = 78848 halves = 157696 bytes (< 320KB WGP LDS)
// ---------------------------------------------------------------------------
#define CONV_LDS_BYTES (size_t)((65536 + 4096 + 8192 + 1024) * 2)

__global__ __launch_bounds__(256) void k_conv(
    const int* __restrict__ obs, const float* __restrict__ max_vec,
    const _Float16* __restrict__ w1h, const float* __restrict__ b1,
    const _Float16* __restrict__ w2h, const float* __restrict__ b2,
    _Float16* __restrict__ feats) {
  extern __shared__ _Float16 smem[];
  _Float16* box   = smem;                    // [8][32][16][16]
  _Float16* apan  = box + IMGS * NL * 256;   // [128][32]
  _Float16* c1out = apan + 128 * 32;         // [128][64]
  _Float16* a2pan = c1out + 128 * 64;        // [32][32]

  const int tid  = threadIdx.x;
  const int wave = tid >> 5;
  const int lane = tid & 31;
  const int col  = lane & 15;
  const int rh   = (lane >> 4) * 8;
  const int imgbase = blockIdx.x * IMGS;

  // -- zero the box (u32 stores) --------------------------------------------
  {
    uint32_t* bw = (uint32_t*)box;
#pragma unroll 4
    for (int i = tid; i < IMGS * NL * 128; i += 256) bw[i] = 0u;
  }
  __syncthreads();

  // -- scatter tokens, normalized by max_vec --------------------------------
  for (int t = tid; t < IMGS * NTOK; t += 256) {
    const int img = t / NTOK;
    const int m   = t - img * NTOK;
    const int* o  = obs + ((size_t)(imgbase + img) * NTOK + m) * 3;
    const int coord = o[0], attr = o[1], val = o[2];
    if (coord != 255 && attr < NL) {
      const int x = (coord >> 4) & 15, y = coord & 15;
      const float f = (float)val / max_vec[attr];
      box[img * (NL * 256) + attr * 256 + x * 16 + y] = (_Float16)f;
    }
  }
  __syncthreads();

  // -- conv1 as GEMM: M=128 (img*16+pos), N=64, K=800 -----------------------
  v8f acc[4];
#pragma unroll
  for (int i = 0; i < 4; ++i) acc[i] = vzero8();
  const int mtile = wave;   // wave == image
  for (int ks = 0; ks < 25; ++ks) {
    const int k0 = ks * 32;
    // cooperative im2col panel fill: 128x32
    for (int i = tid; i < 128 * 32; i += 256) {
      const int row = i >> 5, kk = i & 31;
      const int kg = k0 + kk;
      const int ch = kg / 25, r5 = kg - ch * 25;
      const int ky = r5 / 5, kx = r5 - ky * 5;
      const int img = row >> 4, pos = row & 15;
      const int oh = pos >> 2, ow = pos & 3;
      const int h = oh * 3 + ky, w = ow * 3 + kx;
      apan[i] = box[img * (NL * 256) + ch * 256 + h * 16 + w];
    }
    __syncthreads();
    const v16h a = load_frag16(apan + mtile * 16 * 32, 32);
#pragma unroll
    for (int nt = 0; nt < 4; ++nt) {
      const v16h b = load_frag16(w1h + (nt * 16) * K1LEN + k0, K1LEN);
      acc[nt] = wmma16(a, b, acc[nt]);
    }
    __syncthreads();
  }
  // bias + relu -> c1out f16
#pragma unroll
  for (int nt = 0; nt < 4; ++nt) {
    const float bb = b1[nt * 16 + col];
#pragma unroll
    for (int j = 0; j < 8; ++j) {
      float v = acc[nt][j] + bb;
      v = v > 0.0f ? v : 0.0f;
      c1out[(mtile * 16 + rh + j) * 64 + nt * 16 + col] = (_Float16)v;
    }
  }
  __syncthreads();

  // -- conv2 as GEMM: M=32 (img*4+pos), N=64, K=576 -------------------------
  v8f acc2 = vzero8();
  const int mt2 = wave >> 2;   // 0..1
  const int nt2 = wave & 3;    // 0..3
  for (int ks = 0; ks < 18; ++ks) {
    const int k0 = ks * 32;
    for (int i = tid; i < 32 * 32; i += 256) {
      const int row = i >> 5, kk = i & 31;
      const int kg = k0 + kk;
      const int ch = kg / 9, r3 = kg - ch * 9;
      const int ky = r3 / 3, kx = r3 - ky * 3;
      const int img = row >> 2, pos = row & 3;
      const int oh = pos >> 1, ow = pos & 1;
      const int h = oh + ky, w = ow + kx;
      a2pan[i] = c1out[(img * 16 + h * 4 + w) * 64 + ch];
    }
    __syncthreads();
    const v16h a = load_frag16(a2pan + mt2 * 16 * 32, 32);
    const v16h b = load_frag16(w2h + (nt2 * 16) * K2LEN + k0, K2LEN);
    acc2 = wmma16(a, b, acc2);
    __syncthreads();
  }
  // bias + relu -> flat features [B][256], flat index = ch*4 + (oh*2+ow)
  {
    const int ch = nt2 * 16 + col;
    const float bb = b2[ch];
#pragma unroll
    for (int j = 0; j < 8; ++j) {
      const int row = mt2 * 16 + rh + j;
      const int img = row >> 2, pos = row & 3;
      float v = acc2[j] + bb;
      v = v > 0.0f ? v : 0.0f;
      feats[((size_t)(imgbase + img)) * FLATW + ch * 4 + pos] = (_Float16)v;
    }
  }
}

// ---------------------------------------------------------------------------
// K2: fc1 -> enc -> critic/value -> actor1 -> query -> logits
// 16 batch rows per workgroup, 8 waves.
// ---------------------------------------------------------------------------
__global__ __launch_bounds__(256) void k_mlp(
    const _Float16* __restrict__ feats,
    const _Float16* __restrict__ fc1h, const float* __restrict__ fc1b,
    const _Float16* __restrict__ ench, const float* __restrict__ encb,
    const _Float16* __restrict__ crith, const float* __restrict__ critb,
    const float* __restrict__ valw, const float* __restrict__ valb,
    const _Float16* __restrict__ act1h, const float* __restrict__ act1b,
    const _Float16* __restrict__ actWth,
    const _Float16* __restrict__ embh, const float* __restrict__ abias,
    float* __restrict__ logits, float* __restrict__ value) {
  __shared__ _Float16 xin[16 * 256];
  __shared__ _Float16 h1[16 * 128];
  __shared__ _Float16 hid[16 * 128];
  __shared__ _Float16 af[16 * 512];
  __shared__ _Float16 qpan[16 * 32];
  __shared__ float valacc[16];

  const int tid  = threadIdx.x;
  const int wave = tid >> 5;
  const int lane = tid & 31;
  const int col  = lane & 15;
  const int rh   = (lane >> 4) * 8;
  const size_t rowbase = (size_t)blockIdx.x * 16;

  // load 16x256 f16 features (as dwords)
  {
    const uint32_t* src = (const uint32_t*)(feats + rowbase * FLATW);
    uint32_t* dst = (uint32_t*)xin;
#pragma unroll 4
    for (int i = tid; i < 16 * 128; i += 256) dst[i] = src[i];
  }
  if (tid < 16) valacc[tid] = 0.0f;
  __syncthreads();

  // -- fc1: M16 N128 K256 (wave -> N-tile) ----------------------------------
  {
    v8f acc = vzero8();
#pragma unroll
    for (int ks = 0; ks < 8; ++ks) {
      const v16h a = load_frag16(xin + ks * 32, 256);
      const v16h b = load_frag16(fc1h + (wave * 16) * 256 + ks * 32, 256);
      acc = wmma16(a, b, acc);
    }
    const float bb = fc1b[wave * 16 + col];
#pragma unroll
    for (int j = 0; j < 8; ++j) {
      float v = acc[j] + bb;
      v = v > 0.0f ? v : 0.0f;
      h1[(rh + j) * 128 + wave * 16 + col] = (_Float16)v;
    }
  }
  __syncthreads();

  // -- enc: M16 N128 K128 ---------------------------------------------------
  {
    v8f acc = vzero8();
#pragma unroll
    for (int ks = 0; ks < 4; ++ks) {
      const v16h a = load_frag16(h1 + ks * 32, 128);
      const v16h b = load_frag16(ench + (wave * 16) * 128 + ks * 32, 128);
      acc = wmma16(a, b, acc);
    }
    const float bb = encb[wave * 16 + col];
#pragma unroll
    for (int j = 0; j < 8; ++j) {
      float v = acc[j] + bb;
      v = v > 0.0f ? v : 0.0f;
      hid[(rh + j) * 128 + wave * 16 + col] = (_Float16)v;
    }
  }
  __syncthreads();

  // -- critic: M16 N1024 K128; tanh + dot(value_w) reduced in LDS -----------
  {
    float part[8];
#pragma unroll
    for (int j = 0; j < 8; ++j) part[j] = 0.0f;
    for (int t = 0; t < 8; ++t) {
      const int nt = wave * 8 + t;
      v8f acc = vzero8();
#pragma unroll
      for (int ks = 0; ks < 4; ++ks) {
        const v16h a = load_frag16(hid + ks * 32, 128);
        const v16h b = load_frag16(crith + (nt * 16) * 128 + ks * 32, 128);
        acc = wmma16(a, b, acc);
      }
      const int n = nt * 16 + col;
      const float bb = critb[n];
      const float vw = valw[n];
#pragma unroll
      for (int j = 0; j < 8; ++j) part[j] += tanhf(acc[j] + bb) * vw;
    }
#pragma unroll
    for (int j = 0; j < 8; ++j) atomicAdd(&valacc[rh + j], part[j]);
  }

  // -- actor1: M16 N512 K128 ------------------------------------------------
  for (int t = 0; t < 4; ++t) {
    const int nt = wave * 4 + t;
    v8f acc = vzero8();
#pragma unroll
    for (int ks = 0; ks < 4; ++ks) {
      const v16h a = load_frag16(hid + ks * 32, 128);
      const v16h b = load_frag16(act1h + (nt * 16) * 128 + ks * 32, 128);
      acc = wmma16(a, b, acc);
    }
    const float bb = act1b[nt * 16 + col];
#pragma unroll
    for (int j = 0; j < 8; ++j) {
      float v = acc[j] + bb;
      v = v > 0.0f ? v : 0.0f;
      af[(rh + j) * 512 + nt * 16 + col] = (_Float16)v;
    }
  }
  __syncthreads();

  // value output (critic atomics complete after barrier above)
  if (tid < 16) value[rowbase + tid] = valacc[tid] + valb[0];

  // -- query: M16 N16 K512, tanh; zero-pad K panel to 32 for logits step ----
  if (wave == 0) {
    v8f acc = vzero8();
    for (int ks = 0; ks < 16; ++ks) {
      const v16h a = load_frag16(af + ks * 32, 512);
      const v16h b = load_frag16(actWth + ks * 32, 512);
      acc = wmma16(a, b, acc);
    }
#pragma unroll
    for (int j = 0; j < 8; ++j) {
      qpan[(rh + j) * 32 + col]      = (_Float16)tanhf(acc[j]);
      qpan[(rh + j) * 32 + 16 + col] = (_Float16)0.0f;
    }
  }
  __syncthreads();

  // -- logits: M16 N100(pad 112) K16(pad 32); waves 0..6 --------------------
  if (wave < 7) {
    const v16h a = load_frag16(qpan, 32);
    const v16h b = load_frag16(embh + (wave * 16) * 32, 32);
    v8f acc = vzero8();
    acc = wmma16(a, b, acc);
    const float ab = abias[0];
    const int n = wave * 16 + col;
    if (n < NACT) {
#pragma unroll
      for (int j = 0; j < 8; ++j)
        logits[(rowbase + rh + j) * NACT + n] = acc[j] + ab;
    }
  }
}

// ---------------------------------------------------------------------------
// Launch
// ---------------------------------------------------------------------------
extern "C" void kernel_launch(void* const* d_in, const int* in_sizes, int n_in,
                              void* d_out, int out_size, void* d_ws,
                              size_t ws_size, hipStream_t stream) {
  (void)in_sizes; (void)n_in; (void)out_size; (void)ws_size;
  const int*   obs     = (const int*)d_in[0];
  const float* max_vec = (const float*)d_in[1];
  const float* cnn1_w  = (const float*)d_in[2];
  const float* cnn1_b  = (const float*)d_in[3];
  const float* cnn2_w  = (const float*)d_in[4];
  const float* cnn2_b  = (const float*)d_in[5];
  const float* fc1_w   = (const float*)d_in[6];
  const float* fc1_b   = (const float*)d_in[7];
  const float* enc_w   = (const float*)d_in[8];
  const float* enc_b   = (const float*)d_in[9];
  const float* crit_w  = (const float*)d_in[10];
  const float* crit_b  = (const float*)d_in[11];
  const float* val_w   = (const float*)d_in[12];
  const float* val_b   = (const float*)d_in[13];
  const float* act1_w  = (const float*)d_in[14];
  const float* act1_b  = (const float*)d_in[15];
  const float* emb     = (const float*)d_in[16];
  const float* actW    = (const float*)d_in[17];
  const float* abias   = (const float*)d_in[18];

  _Float16* ws = (_Float16*)d_ws;
  float* logits = (float*)d_out;                 // [8192][100]
  float* value  = (float*)d_out + (size_t)NB * NACT;  // [8192]

  k_prep<<<256, 256, 0, stream>>>(cnn1_w, cnn2_w, fc1_w, enc_w, crit_w,
                                  act1_w, actW, emb, ws);

  (void)hipFuncSetAttribute((const void*)k_conv,
                            hipFuncAttributeMaxDynamicSharedMemorySize,
                            (int)CONV_LDS_BYTES);
  k_conv<<<NB / IMGS, 256, CONV_LDS_BYTES, stream>>>(
      obs, max_vec, ws + OFF_W1, cnn1_b, ws + OFF_W2, cnn2_b, ws + OFF_FEATS);

  k_mlp<<<NB / 16, 256, 0, stream>>>(
      ws + OFF_FEATS, ws + OFF_FC1, fc1_b, ws + OFF_ENC, enc_b, ws + OFF_CRIT,
      crit_b, val_w, val_b, ws + OFF_ACT1, act1_b, ws + OFF_ACTWT,
      ws + OFF_EMB, abias, logits, value);
}
